// ViT_5497558139639
// MI455X (gfx1250) — matrix operand order repaired and checked by reading, not compiled, not run on hardware
//
#include <hip/hip_runtime.h>
#include <math.h>
#include <stdint.h>

typedef __attribute__((ext_vector_type(16))) _Float16 v16h;
typedef __attribute__((ext_vector_type(8)))  float    v8f;

#define B_    4
#define SEQ_  577
#define NTOK_ 576
#define E_    384
#define H_    6
#define DH_   64
#define MLP_  1536
#define TOKD_ 768
#define IMG_  384
#define P_    16
#define NBLK_ 12
#define SLD_  608   // padded leading dim for score rows: 19*32, rows 16B-aligned

#define LDA_T 36    // LDS stride (floats) for A tile rows   (16B-aligned b128 dsts)
#define LDB_T 68    // LDS stride (floats) for B tile rows (K x N layout)
#define LDBT_T 36   // LDS stride (floats) for B^T tile rows (N x K layout)

// ---- CDNA5 async global->LDS copy (ASYNCcnt-tracked), ISA 15.18.3 op 98 ----
__device__ __forceinline__ void async_copy_b128(unsigned lds_byte_addr, const void* gaddr)
{
    asm volatile("global_load_async_to_lds_b128 %0, %1, off"
                 :: "v"(lds_byte_addr), "v"(gaddr)
                 : "memory");
}
__device__ __forceinline__ void wait_async0()
{
    asm volatile("s_wait_asynccnt 0x0" ::: "memory");
}
__device__ __forceinline__ unsigned lds_addr(const void* p)
{
    // generic shared address: low 32 bits are the LDS byte offset (ISA 10.2)
    return (unsigned)(uintptr_t)p;
}

// ---------------------------------------------------------------------------
// Generic batched GEMM via WMMA:  C = act(scale*(A@B) + bias) + Res
//  A: row-major M x K (lda), batch stride aBatch
//  B: bT==0 -> row-major K x N (ldb);  bT==1 -> row-major N x K (B^T supplied)
// Block: 256 threads = 8 waves; tile 64(M) x 64(N); K step 32.
// Waves 4(M) x 2(N); each wave: 16x32 via 2x v_wmma_f32_16x16x32_f16.
// Interior tiles: double-buffered GLOBAL_LOAD_ASYNC_TO_LDS_B128 pipeline
// (issue next K-tile, run WMMAs on current, s_wait_asynccnt + barrier).
// Edge tiles: synchronous clamped loads + selects.
// ---------------------------------------------------------------------------
__launch_bounds__(256)
__global__ void gemm_wmma_kernel(const float* __restrict__ A, long long aBatch, int lda,
                                 const float* __restrict__ Bm, long long bBatch, int ldb, int bT,
                                 const float* __restrict__ bias,
                                 const float* __restrict__ Res, long long rBatch,
                                 float* __restrict__ C, long long cBatch, int ldc,
                                 int M, int N, int K, float scale, int act)
{
    __shared__ __attribute__((aligned(16))) float lAbuf[2][64 * LDA_T];
    __shared__ __attribute__((aligned(16))) float lBbuf[2][64 * LDBT_T]; // fits 32*LDB_T too

    const int bz = blockIdx.z;
    A  += (long long)bz * aBatch;
    Bm += (long long)bz * bBatch;
    C  += (long long)bz * cBatch;
    const float* Rp = Res ? (Res + (long long)bz * rBatch) : (const float*)0;

    const int m0   = blockIdx.y * 64;
    const int n0   = blockIdx.x * 64;
    const int tid  = threadIdx.x;
    const int lane = tid & 31;
    const int wave = tid >> 5;
    const int g    = lane >> 4;        // half-wave group
    const int l15  = lane & 15;
    const int mw   = (wave >> 1) * 16; // wave M offset in tile (0/16/32/48)
    const int nw   = (wave & 1) * 32;  // wave N offset in tile (0/32)

    // staging thread->element maps (8 floats = 2 x b128 per thread per matrix)
    const int ar = tid >> 2, ac = (tid & 3) * 8;   // A: 64 rows x 32 cols
    const int br = tid >> 3, bc = (tid & 7) * 8;   // B (KxN): 32 rows x 64 cols
    const int bn = tid >> 2, bk = (tid & 3) * 8;   // B^T (NxK): 64 rows x 32 cols

    v8f acc[2] = {{}, {}};

    const bool interior = (m0 + 64 <= M) && (n0 + 64 <= N) && ((K & 31) == 0);

    if (interior) {
        const float* Ap = A + (long long)(m0 + ar) * lda + ac;
        // prologue: async-stage first K tile into buffer 0
        {
            float* dA = lAbuf[0];
            float* dB = lBbuf[0];
            async_copy_b128(lds_addr(dA + ar * LDA_T + ac),     Ap);
            async_copy_b128(lds_addr(dA + ar * LDA_T + ac + 4), Ap + 4);
            if (!bT) {
                const float* Bp = Bm + (long long)br * ldb + n0 + bc;
                async_copy_b128(lds_addr(dB + br * LDB_T + bc),     Bp);
                async_copy_b128(lds_addr(dB + br * LDB_T + bc + 4), Bp + 4);
            } else {
                const float* Bp = Bm + (long long)(n0 + bn) * ldb + bk;
                async_copy_b128(lds_addr(dB + bn * LDBT_T + bk),     Bp);
                async_copy_b128(lds_addr(dB + bn * LDBT_T + bk + 4), Bp + 4);
            }
            wait_async0();
            __syncthreads();
        }
        for (int k0 = 0; k0 < K; k0 += 32) {
            const int cur = (k0 >> 5) & 1;
            const float* cA = lAbuf[cur];
            const float* cB = lBbuf[cur];
            const bool more = (k0 + 32) < K;
            if (more) {  // async-stage next K tile into the other buffer
                float* dA = lAbuf[cur ^ 1];
                float* dB = lBbuf[cur ^ 1];
                async_copy_b128(lds_addr(dA + ar * LDA_T + ac),     Ap + k0 + 32);
                async_copy_b128(lds_addr(dA + ar * LDA_T + ac + 4), Ap + k0 + 36);
                if (!bT) {
                    const float* Bp = Bm + (long long)(k0 + 32 + br) * ldb + n0 + bc;
                    async_copy_b128(lds_addr(dB + br * LDB_T + bc),     Bp);
                    async_copy_b128(lds_addr(dB + br * LDB_T + bc + 4), Bp + 4);
                } else {
                    const float* Bp = Bm + (long long)(n0 + bn) * ldb + k0 + 32 + bk;
                    async_copy_b128(lds_addr(dB + bn * LDBT_T + bk),     Bp);
                    async_copy_b128(lds_addr(dB + bn * LDBT_T + bk + 4), Bp + 4);
                }
            }
            // ---- fragments + WMMA on current buffer ----
            v16h af;
#pragma unroll
            for (int e = 0; e < 16; ++e) {
                int ka = e + 16 * g;                      // contiguous k = 16*g + e, same order as B
                af[e] = (_Float16)cA[(mw + l15) * LDA_T + ka];
            }
#pragma unroll
            for (int t = 0; t < 2; ++t) {
                v16h bf;
                if (!bT) {
#pragma unroll
                    for (int e = 0; e < 16; ++e) {
                        int kb = e + 16 * g;              // contiguous k = 16*g + e, same order as A
                        bf[e] = (_Float16)cB[kb * LDB_T + nw + t * 16 + l15];
                    }
                } else {
#pragma unroll
                    for (int e = 0; e < 16; ++e) {
                        int kb = e + 16 * g;
                        bf[e] = (_Float16)cB[(nw + t * 16 + l15) * LDBT_T + kb];
                    }
                }
                acc[t] = __builtin_amdgcn_wmma_f32_16x16x32_f16(false, af, false, bf,
                                                                (short)0, acc[t], false, false);
            }
            if (more) wait_async0();
            __syncthreads();
        }
    } else {
        float* lA = lAbuf[0];
        float* lB = lBbuf[0];
        const int gmc = (m0 + ar < M) ? (m0 + ar) : (M - 1);
        const bool mok = (m0 + ar) < M;
        for (int k0 = 0; k0 < K; k0 += 32) {
            // ---- clamped loads + select (no OOB addresses) ----
#pragma unroll
            for (int j = 0; j < 8; ++j) {
                int gk = k0 + ac + j;
                int gkc = (gk < K) ? gk : (K - 1);
                float t = A[(long long)gmc * lda + gkc];
                lA[ar * LDA_T + ac + j] = (mok && gk < K) ? t : 0.0f;
            }
            if (!bT) {
                int gk = k0 + br;
                int gkc = (gk < K) ? gk : (K - 1);
                bool kok = gk < K;
#pragma unroll
                for (int j = 0; j < 8; ++j) {
                    int gn = n0 + bc + j;
                    int gnc = (gn < N) ? gn : (N - 1);
                    float t = Bm[(long long)gkc * ldb + gnc];
                    lB[br * LDB_T + bc + j] = (kok && gn < N) ? t : 0.0f;
                }
            } else {
                int gn = n0 + bn;
                int gnc = (gn < N) ? gn : (N - 1);
                bool nok = gn < N;
#pragma unroll
                for (int j = 0; j < 8; ++j) {
                    int gk = k0 + bk + j;
                    int gkc = (gk < K) ? gk : (K - 1);
                    float t = Bm[(long long)gnc * ldb + gkc];
                    lB[bn * LDBT_T + bk + j] = (nok && gk < K) ? t : 0.0f;
                }
            }
            __syncthreads();

            v16h af;
#pragma unroll
            for (int e = 0; e < 16; ++e) {
                int ka = e + 16 * g;
                af[e] = (_Float16)lA[(mw + l15) * LDA_T + ka];
            }
#pragma unroll
            for (int t = 0; t < 2; ++t) {
                v16h bf;
                if (!bT) {
#pragma unroll
                    for (int e = 0; e < 16; ++e) {
                        int kb = e + 16 * g;
                        bf[e] = (_Float16)lB[kb * LDB_T + nw + t * 16 + l15];
                    }
                } else {
#pragma unroll
                    for (int e = 0; e < 16; ++e) {
                        int kb = e + 16 * g;
                        bf[e] = (_Float16)lB[(nw + t * 16 + l15) * LDBT_T + kb];
                    }
                }
                acc[t] = __builtin_amdgcn_wmma_f32_16x16x32_f16(false, af, false, bf,
                                                                (short)0, acc[t], false, false);
            }
            __syncthreads();
        }
    }

    // ---- epilogue: scale, bias, gelu, residual ----
    const bool storeFast = (m0 + 64 <= M) && (n0 + 64 <= N);
#pragma unroll
    for (int t = 0; t < 2; ++t) {
        int col = n0 + nw + t * 16 + l15;
#pragma unroll
        for (int r = 0; r < 8; ++r) {
            int row = m0 + mw + r + 8 * g;
            if (storeFast || (row < M && col < N)) {
                float v = acc[t][r] * scale;
                if (bias) v += bias[col];
                if (act == 1) v = 0.5f * v * (1.0f + erff(v * 0.70710678118654752f));
                if (Rp) v += Rp[(long long)row * ldc + col];
                C[(long long)row * ldc + col] = v;
            }
        }
    }
}

// ---------------------------------------------------------------------------
// LayerNorm over E_=384, one 128-thread block per row.
// ---------------------------------------------------------------------------
__launch_bounds__(128)
__global__ void ln_kernel(const float* __restrict__ x, const float* __restrict__ gm,
                          const float* __restrict__ bt, float* __restrict__ out)
{
    __shared__ float red[128];
    long long row = blockIdx.x;
    const float* xr = x + row * E_;
    float* orow = out + row * E_;
    int tid = threadIdx.x;
    float v0 = xr[tid], v1 = xr[tid + 128], v2 = xr[tid + 256];
    red[tid] = v0 + v1 + v2;
    __syncthreads();
    for (int off = 64; off > 0; off >>= 1) { if (tid < off) red[tid] += red[tid + off]; __syncthreads(); }
    float mu = red[0] * (1.0f / E_);
    __syncthreads();
    float d0 = v0 - mu, d1 = v1 - mu, d2 = v2 - mu;
    red[tid] = d0 * d0 + d1 * d1 + d2 * d2;
    __syncthreads();
    for (int off = 64; off > 0; off >>= 1) { if (tid < off) red[tid] += red[tid + off]; __syncthreads(); }
    float rstd = rsqrtf(red[0] * (1.0f / E_) + 1e-5f);
    orow[tid]       = d0 * rstd * gm[tid]       + bt[tid];
    orow[tid + 128] = d1 * rstd * gm[tid + 128] + bt[tid + 128];
    orow[tid + 256] = d2 * rstd * gm[tid + 256] + bt[tid + 256];
}

// ---------------------------------------------------------------------------
// Row softmax over SEQ_=577 with row stride SLD_ (in place). Zeroes the pad
// columns [577, 608) so the P*V GEMM can run K=608 unguarded.
// ---------------------------------------------------------------------------
__launch_bounds__(128)
__global__ void softmax_kernel(float* __restrict__ s)
{
    __shared__ float red[128];
    long long row = blockIdx.x;
    float* r = s + row * (long long)SLD_;
    int tid = threadIdx.x;
    float m = -3.0e38f;
    for (int i = tid; i < SEQ_; i += 128) m = fmaxf(m, r[i]);
    red[tid] = m;
    __syncthreads();
    for (int off = 64; off > 0; off >>= 1) { if (tid < off) red[tid] = fmaxf(red[tid], red[tid + off]); __syncthreads(); }
    m = red[0];
    __syncthreads();
    float sum = 0.0f;
    for (int i = tid; i < SEQ_; i += 128) { float e = expf(r[i] - m); r[i] = e; sum += e; }
    red[tid] = sum;
    __syncthreads();
    for (int off = 64; off > 0; off >>= 1) { if (tid < off) red[tid] += red[tid + off]; __syncthreads(); }
    float inv = 1.0f / red[0];
    for (int i = tid; i < SEQ_; i += 128) r[i] *= inv;
    for (int i = SEQ_ + tid; i < SLD_; i += 128) r[i] = 0.0f;   // zero K-pad
}

// ---------------------------------------------------------------------------
// patchify: out[b, n=x*24+y, e=p1*48+p2*3+c] = img[b, c, p1*24+x, p2*24+y]
//   ('b c (p x) (p y)' with p the OUTER factor -> pixel stride 24 in image)
// ---------------------------------------------------------------------------
__global__ void patchify_kernel(const float* __restrict__ img, float* __restrict__ out)
{
    long long i = (long long)blockIdx.x * 256 + threadIdx.x;
    const long long total = (long long)B_ * NTOK_ * TOKD_;
    if (i >= total) return;
    int e = (int)(i % TOKD_);
    long long r = i / TOKD_;
    int n = (int)(r % NTOK_);
    int b = (int)(r / NTOK_);
    int c  = e % 3;
    int p2 = (e / 3) % P_;
    int p1 = e / 48;
    int yq = n % 24;
    int xq = n / 24;
    int h = p1 * 24 + xq;
    int w = p2 * 24 + yq;
    out[i] = img[(((long long)b * 3 + c) * IMG_ + h) * IMG_ + w];
}

// cls token + positional embedding (xs rows 1.. already hold patch projections)
__global__ void embed_kernel(float* __restrict__ xs, const float* __restrict__ cls,
                             const float* __restrict__ pos)
{
    long long i = (long long)blockIdx.x * 256 + threadIdx.x;
    const long long total = (long long)B_ * SEQ_ * E_;
    if (i >= total) return;
    int e = (int)(i % E_);
    int t = (int)((i / E_) % SEQ_);
    float v = (t == 0) ? cls[e] : xs[i];
    xs[i] = v + pos[(long long)t * E_ + e];
}

// qkv (B,SEQ,3E with [d*18 + k*6 + h] layout) -> Q/K/V (B,H,SEQ,64); any may be null
__global__ void split_qkv_kernel(const float* __restrict__ qkv,
                                 float* __restrict__ Q, float* __restrict__ K,
                                 float* __restrict__ V)
{
    long long i = (long long)blockIdx.x * 256 + threadIdx.x;
    const long long total = (long long)B_ * H_ * SEQ_ * DH_;
    if (i >= total) return;
    int d = (int)(i & 63);
    long long r = i >> 6;
    int t = (int)(r % SEQ_); r /= SEQ_;
    int h = (int)(r % H_);
    int b = (int)(r / H_);
    long long src = ((long long)b * SEQ_ + t) * (3 * E_) + d * (3 * H_) + h;
    if (Q) Q[i] = qkv[src];
    if (K) K[i] = qkv[src + H_];
    if (V) V[i] = qkv[src + 2 * H_];
}

// O (B,H,SEQ,64) -> merged (B,SEQ,E) with e = h*64 + d
__global__ void merge_heads_kernel(const float* __restrict__ O, float* __restrict__ out)
{
    long long i = (long long)blockIdx.x * 256 + threadIdx.x;
    const long long total = (long long)B_ * SEQ_ * E_;
    if (i >= total) return;
    int e = (int)(i % E_);
    long long r = i / E_;
    int t = (int)(r % SEQ_);
    int b = (int)(r / SEQ_);
    int h = e >> 6;
    int d = e & 63;
    out[i] = O[(((long long)(b * H_ + h)) * SEQ_ + t) * DH_ + d];
}

// final slice: out[b, t-1, :] = xs[b, t, :]  (drop cls token)
__global__ void copy_out_kernel(const float* __restrict__ xs, float* __restrict__ out)
{
    long long i = (long long)blockIdx.x * 256 + threadIdx.x;
    const long long total = (long long)B_ * NTOK_ * E_;
    if (i >= total) return;
    int e = (int)(i % E_);
    long long r = i / E_;
    int n = (int)(r % NTOK_);
    int b = (int)(r / NTOK_);
    out[i] = xs[((long long)b * SEQ_ + n + 1) * E_ + e];
}

// ---------------------------------------------------------------------------
// Host side
// ---------------------------------------------------------------------------
static inline void launch_gemm(hipStream_t st,
                               const float* A, long long aB, int lda,
                               const float* Bm, long long bB, int ldb, int bT,
                               const float* bias, const float* Res, long long rB,
                               float* C, long long cB, int ldc,
                               int M, int N, int K, float scale, int act, int batches)
{
    dim3 grid((N + 63) / 64, (M + 63) / 64, batches);
    gemm_wmma_kernel<<<grid, 256, 0, st>>>(A, aB, lda, Bm, bB, ldb, bT,
                                           bias, Res, rB, C, cB, ldc,
                                           M, N, K, scale, act);
}

static inline unsigned ew_grid(long long n) { return (unsigned)((n + 255) / 256); }

extern "C" void kernel_launch(void* const* d_in, const int* in_sizes, int n_in,
                              void* d_out, int out_size, void* d_ws, size_t ws_size,
                              hipStream_t stream)
{
    (void)in_sizes; (void)n_in; (void)out_size; (void)ws_size;

    const float* x_img  = (const float*)d_in[0];
    const float* y_img  = (const float*)d_in[1];
    const float* proj1w = (const float*)d_in[2];
    const float* proj1b = (const float*)d_in[3];
    const float* proj2w = (const float*)d_in[4];
    const float* proj2b = (const float*)d_in[5];
    const float* cls    = (const float*)d_in[6];
    const float* pos    = (const float*)d_in[7];
    // blocks dict order: ln1_g, ln1_b, ..., ln6_g, ln6_b (8..19), then weights (20..30)
    const float* lnP[12];
    for (int i = 0; i < 12; ++i) lnP[i] = (const float*)d_in[8 + i];
    const float* a1_qkv  = (const float*)d_in[20];
    const float* a1_out  = (const float*)d_in[21];
    const float* a2_qkv  = (const float*)d_in[22];
    const float* a2_out  = (const float*)d_in[23];
    const float* cx_qkv1 = (const float*)d_in[24];
    const float* cx_qkv2 = (const float*)d_in[25];
    const float* cx_out  = (const float*)d_in[26];
    const float* mlp_w1  = (const float*)d_in[27];
    const float* mlp_w2  = (const float*)d_in[28];
    const float* mlp_b1  = (const float*)d_in[29];
    const float* mlp_b2  = (const float*)d_in[30];

    const long long QKV_STRIDE = (long long)E_ * 3 * E_;   // 442368
    const long long OUT_STRIDE = (long long)E_ * E_;       // 147456
    const long long MLP_STRIDE = (long long)E_ * MLP_;     // 589824

    // ---- workspace layout (floats; every buffer 16B-aligned) ----
    float* ws = (float*)d_ws;
    long long off = 0;
    float* patch  = ws + off; off += (long long)B_ * NTOK_ * TOKD_;
    float* xs     = ws + off; off += (long long)B_ * SEQ_ * E_;
    float* ys     = ws + off; off += (long long)B_ * SEQ_ * E_;
    float* xn     = ws + off; off += (long long)B_ * SEQ_ * E_;
    float* yn     = ws + off; off += (long long)B_ * SEQ_ * E_;
    float* qkvX   = ws + off; off += (long long)B_ * SEQ_ * 3 * E_;
    float* qkvY   = ws + off; off += (long long)B_ * SEQ_ * 3 * E_;
    float* Qh     = ws + off; off += (long long)B_ * H_ * SEQ_ * DH_;
    float* Kh     = ws + off; off += (long long)B_ * H_ * SEQ_ * DH_;
    float* Vh     = ws + off; off += (long long)B_ * H_ * SEQ_ * DH_;
    float* scores = ws + off; off += (long long)B_ * H_ * SEQ_ * SLD_;
    float* Obuf   = ws + off; off += (long long)B_ * H_ * SEQ_ * DH_;
    float* merged = ws + off; off += (long long)B_ * SEQ_ * E_;
    float* x2     = ws + off; off += (long long)B_ * SEQ_ * E_;
    float* y2     = ws + off; off += (long long)B_ * SEQ_ * E_;
    float* y3     = ws + off; off += (long long)B_ * SEQ_ * E_;
    float* mlph   = ws + off; off += (long long)B_ * SEQ_ * MLP_;

    const long long rowsBS = (long long)B_ * SEQ_;           // 2308
    const long long nBSE   = rowsBS * E_;
    const int BH = B_ * H_;                                  // 24
    const long long SCB = (long long)SEQ_ * SLD_;            // score batch stride
    const float invdk_self  = 1.0f / 32.0f;                  // (E/H)/2, the torch "bug"
    const float invdk_cross = 1.0f / 8.0f;                   // sqrt(E/H)

    // ---- patch embeddings + cls + pos ----
    patchify_kernel<<<ew_grid((long long)B_ * NTOK_ * TOKD_), 256, 0, stream>>>(x_img, patch);
    launch_gemm(stream, patch, (long long)NTOK_ * TOKD_, TOKD_,
                proj1w, 0, E_, 0, proj1b, nullptr, 0,
                xs + E_, (long long)SEQ_ * E_, E_, NTOK_, E_, TOKD_, 1.0f, 0, B_);
    patchify_kernel<<<ew_grid((long long)B_ * NTOK_ * TOKD_), 256, 0, stream>>>(y_img, patch);
    launch_gemm(stream, patch, (long long)NTOK_ * TOKD_, TOKD_,
                proj2w, 0, E_, 0, proj2b, nullptr, 0,
                ys + E_, (long long)SEQ_ * E_, E_, NTOK_, E_, TOKD_, 1.0f, 0, B_);
    embed_kernel<<<ew_grid(nBSE), 256, 0, stream>>>(xs, cls, pos);
    embed_kernel<<<ew_grid(nBSE), 256, 0, stream>>>(ys, cls, pos);
    // ys now holds y0 and stays untouched: every block re-adds the ORIGINAL y.

    for (int blk = 0; blk < NBLK_; ++blk) {
        const float* ln1_g = lnP[0] + blk * E_;  const float* ln1_b = lnP[1] + blk * E_;
        const float* ln2_g = lnP[2] + blk * E_;  const float* ln2_b = lnP[3] + blk * E_;
        const float* ln3_g = lnP[4] + blk * E_;  const float* ln3_b = lnP[5] + blk * E_;
        const float* ln4_g = lnP[6] + blk * E_;  const float* ln4_b = lnP[7] + blk * E_;
        const float* ln5_g = lnP[8] + blk * E_;  const float* ln5_b = lnP[9] + blk * E_;
        const float* w_a1qkv = a1_qkv  + blk * QKV_STRIDE;
        const float* w_a1out = a1_out  + blk * OUT_STRIDE;
        const float* w_a2qkv = a2_qkv  + blk * QKV_STRIDE;
        const float* w_a2out = a2_out  + blk * OUT_STRIDE;
        const float* w_cq1   = cx_qkv1 + blk * QKV_STRIDE;
        const float* w_cq2   = cx_qkv2 + blk * QKV_STRIDE;
        const float* w_cout  = cx_out  + blk * OUT_STRIDE;
        const float* w_m1    = mlp_w1  + blk * MLP_STRIDE;
        const float* w_m2    = mlp_w2  + blk * MLP_STRIDE;
        const float* b_m1    = mlp_b1  + blk * MLP_;
        const float* b_m2    = mlp_b2  + blk * E_;

        // ---- x self-attention:  x2 = x + SA(ln1(x)) ----
        ln_kernel<<<(unsigned)rowsBS, 128, 0, stream>>>(xs, ln1_g, ln1_b, xn);
        launch_gemm(stream, xn, 0, E_, w_a1qkv, 0, 3 * E_, 0, nullptr, nullptr, 0,
                    qkvX, 0, 3 * E_, (int)rowsBS, 3 * E_, E_, 1.0f, 0, 1);
        split_qkv_kernel<<<ew_grid((long long)BH * SEQ_ * DH_), 256, 0, stream>>>(qkvX, Qh, Kh, Vh);
        launch_gemm(stream, Qh, (long long)SEQ_ * DH_, DH_, Kh, (long long)SEQ_ * DH_, DH_, 1,
                    nullptr, nullptr, 0, scores, SCB, SLD_,
                    SEQ_, SEQ_, DH_, invdk_self, 0, BH);
        softmax_kernel<<<(unsigned)(BH * SEQ_), 128, 0, stream>>>(scores);
        launch_gemm(stream, scores, SCB, SLD_, Vh, (long long)SEQ_ * DH_, DH_, 0,
                    nullptr, nullptr, 0, Obuf, (long long)SEQ_ * DH_, DH_,
                    SEQ_, DH_, SLD_, 1.0f, 0, BH);
        merge_heads_kernel<<<ew_grid(nBSE), 256, 0, stream>>>(Obuf, merged);
        launch_gemm(stream, merged, 0, E_, w_a1out, 0, E_, 0, nullptr, xs, 0,
                    x2, 0, E_, (int)rowsBS, E_, E_, 1.0f, 0, 1);

        // ---- y self-attention:  y2 = y0 + SA(ln4(y0)) ----
        ln_kernel<<<(unsigned)rowsBS, 128, 0, stream>>>(ys, ln4_g, ln4_b, yn);
        launch_gemm(stream, yn, 0, E_, w_a2qkv, 0, 3 * E_, 0, nullptr, nullptr, 0,
                    qkvY, 0, 3 * E_, (int)rowsBS, 3 * E_, E_, 1.0f, 0, 1);
        split_qkv_kernel<<<ew_grid((long long)BH * SEQ_ * DH_), 256, 0, stream>>>(qkvY, Qh, Kh, Vh);
        launch_gemm(stream, Qh, (long long)SEQ_ * DH_, DH_, Kh, (long long)SEQ_ * DH_, DH_, 1,
                    nullptr, nullptr, 0, scores, SCB, SLD_,
                    SEQ_, SEQ_, DH_, invdk_self, 0, BH);
        softmax_kernel<<<(unsigned)(BH * SEQ_), 128, 0, stream>>>(scores);
        launch_gemm(stream, scores, SCB, SLD_, Vh, (long long)SEQ_ * DH_, DH_, 0,
                    nullptr, nullptr, 0, Obuf, (long long)SEQ_ * DH_, DH_,
                    SEQ_, DH_, SLD_, 1.0f, 0, BH);
        merge_heads_kernel<<<ew_grid(nBSE), 256, 0, stream>>>(Obuf, merged);
        launch_gemm(stream, merged, 0, E_, w_a2out, 0, E_, 0, nullptr, ys, 0,
                    y2, 0, E_, (int)rowsBS, E_, E_, 1.0f, 0, 1);

        // ---- cross-attention:  y3 = y2 + CA(ln2(x2), ln5(y2)) ----
        ln_kernel<<<(unsigned)rowsBS, 128, 0, stream>>>(x2, ln2_g, ln2_b, xn);
        ln_kernel<<<(unsigned)rowsBS, 128, 0, stream>>>(y2, ln5_g, ln5_b, yn);
        launch_gemm(stream, xn, 0, E_, w_cq1, 0, 3 * E_, 0, nullptr, nullptr, 0,
                    qkvX, 0, 3 * E_, (int)rowsBS, 3 * E_, E_, 1.0f, 0, 1);
        launch_gemm(stream, yn, 0, E_, w_cq2, 0, 3 * E_, 0, nullptr, nullptr, 0,
                    qkvY, 0, 3 * E_, (int)rowsBS, 3 * E_, E_, 1.0f, 0, 1);
        split_qkv_kernel<<<ew_grid((long long)BH * SEQ_ * DH_), 256, 0, stream>>>(qkvX, Qh, nullptr, nullptr);
        split_qkv_kernel<<<ew_grid((long long)BH * SEQ_ * DH_), 256, 0, stream>>>(qkvY, nullptr, Kh, Vh);
        launch_gemm(stream, Qh, (long long)SEQ_ * DH_, DH_, Kh, (long long)SEQ_ * DH_, DH_, 1,
                    nullptr, nullptr, 0, scores, SCB, SLD_,
                    SEQ_, SEQ_, DH_, invdk_cross, 0, BH);
        softmax_kernel<<<(unsigned)(BH * SEQ_), 128, 0, stream>>>(scores);
        launch_gemm(stream, scores, SCB, SLD_, Vh, (long long)SEQ_ * DH_, DH_, 0,
                    nullptr, nullptr, 0, Obuf, (long long)SEQ_ * DH_, DH_,
                    SEQ_, DH_, SLD_, 1.0f, 0, BH);
        merge_heads_kernel<<<ew_grid(nBSE), 256, 0, stream>>>(Obuf, merged);
        launch_gemm(stream, merged, 0, E_, w_cout, 0, E_, 0, nullptr, y2, 0,
                    y3, 0, E_, (int)rowsBS, E_, E_, 1.0f, 0, 1);

        // ---- MLP:  next_x = y3 + W2(gelu(W1(ln3(y3)) + b1)) + b2 ----
        ln_kernel<<<(unsigned)rowsBS, 128, 0, stream>>>(y3, ln3_g, ln3_b, yn);
        launch_gemm(stream, yn, 0, E_, w_m1, 0, MLP_, 0, b_m1, nullptr, 0,
                    mlph, 0, MLP_, (int)rowsBS, MLP_, E_, 1.0f, /*gelu*/1, 1);
        launch_gemm(stream, mlph, 0, MLP_, w_m2, 0, E_, 0, b_m2, y3, 0,
                    xs, 0, E_, (int)rowsBS, E_, MLP_, 1.0f, 0, 1);
        // xs now holds the block output, which is the next block's x.
    }

    copy_out_kernel<<<ew_grid((long long)B_ * NTOK_ * E_), 256, 0, stream>>>(xs, (float*)d_out);
}
